// ChebNetII_84310208020681
// MI455X (gfx1250) — compile-verified, matrix operand order checked by reading, not run on hardware
//
#include <hip/hip_runtime.h>
#include <hip/hip_bf16.h>

typedef __attribute__((ext_vector_type(16))) _Float16 v16h;
typedef __attribute__((ext_vector_type(8)))  _Float16 v8h;
typedef __attribute__((ext_vector_type(4)))  _Float16 v4h;
typedef __attribute__((ext_vector_type(8)))  float    v8f;
typedef __attribute__((ext_vector_type(4)))  float    v4f;

#define NN     100000
#define NFEAT  512
#define NHID   256
#define NOUT   64
#define KORD   10

// LDS row stride for the staged A slab: 512 + 8 pad halves.
// 520 halves = 1040 B = 260 banks = 4*65 banks -> odd multiple of 4 banks,
// so 16 lanes reading 16B from 16 consecutive rows hit disjoint 4-bank groups.
#define ASTRIDE 520

// ---------------- elementwise / setup kernels ----------------

__global__ __launch_bounds__(256) void k_zero(float* __restrict__ p, int n) {
  int i = blockIdx.x * 256 + threadIdx.x;
  if (i < n) p[i] = 0.0f;
}

__global__ __launch_bounds__(256) void k_deg(const int* __restrict__ row,
                                             float* __restrict__ deg, int E) {
  int e = blockIdx.x * 256 + threadIdx.x;
  if (e < E) atomicAdd(&deg[row[e]], 1.0f);
}

__global__ __launch_bounds__(256) void k_dinv(const float* __restrict__ deg,
                                              float* __restrict__ dinv, int n) {
  int i = blockIdx.x * 256 + threadIdx.x;
  if (i < n) {
    float d = deg[i];
    dinv[i] = (d > 0.0f) ? rsqrtf(fmaxf(d, 1.0f)) : 0.0f;
  }
}

__global__ void k_coe(const float* __restrict__ temp, float* __restrict__ coe) {
  int i = threadIdx.x;
  if (i <= KORD) {
    const float pi = 3.14159265358979323846f;
    float s = 0.0f;
    for (int j = 0; j <= KORD; ++j) {
      float t = fmaxf(temp[j], 0.0f);
      float theta = ((float)(KORD - j) + 0.5f) * pi / (float)(KORD + 1);
      s += t * cosf((float)i * theta);
    }
    coe[i] = (2.0f / (float)(KORD + 1)) * s;
  }
}

__global__ __launch_bounds__(256) void k_cvt_f16(const float* __restrict__ in,
                                                 _Float16* __restrict__ out, int n) {
  int i = blockIdx.x * 256 + threadIdx.x;
  if (i < n) out[i] = (_Float16)in[i];
}

__global__ __launch_bounds__(256) void k_negate(float* __restrict__ p, int n) {
  int i = blockIdx.x * 256 + threadIdx.x;
  if (i < n) p[i] = -p[i];
}

__global__ __launch_bounds__(256) void k_combine0(const float* __restrict__ x,
                                                  const float* __restrict__ t1,
                                                  const float* __restrict__ coe,
                                                  float* __restrict__ out, int n) {
  int i = blockIdx.x * 256 + threadIdx.x;
  if (i < n) out[i] = 0.5f * coe[0] * x[i] + coe[1] * t1[i];
}

__global__ __launch_bounds__(256) void k_axpy(float* __restrict__ out,
                                              const float* __restrict__ src,
                                              const float* __restrict__ coe,
                                              int ci, int n) {
  int i = blockIdx.x * 256 + threadIdx.x;
  if (i < n) out[i] += coe[ci] * src[i];
}

// -------- sparse prop: acc[row,:] += scale*(-dinv[row]*dinv[col]) * vin[col,:] --------
// One wave per edge (e is wave-uniform), each lane handles a float2 of the 64 features.

__global__ __launch_bounds__(256) void k_scatter(const int* __restrict__ row,
                                                 const int* __restrict__ col,
                                                 const float* __restrict__ dinv,
                                                 const float* __restrict__ vin,
                                                 float* __restrict__ acc,
                                                 float scale, int E) {
  unsigned idx = blockIdx.x * 256u + threadIdx.x;
  unsigned e = idx >> 5;
  unsigned f = (idx & 31u) * 2u;
  if (e < (unsigned)E) {
    int rr = row[e], cc = col[e];
    float w = -scale * dinv[rr] * dinv[cc];
    const float2 v = *(const float2*)(vin + (size_t)cc * NOUT + f);
    float* dst = acc + (size_t)rr * NOUT + f;
    atomicAdd(dst,     w * v.x);
    atomicAdd(dst + 1, w * v.y);
  }
}

// ---------------- WMMA GEMMs ----------------
// A fragment (16x32 f16): lane r (r<16) holds row m0+r, K = {ka..ka+7, ka+16..ka+23}
// where ka = k0 + 8*(lane>>4). B fragment mirrors this layout indexed by column N,
// i.e. lane r loads W[n0+r][same K set] (B = W^T so B columns = W rows, contiguous).

// GEMM1: block = 8 waves = 32 M-rows x 256 N-cols super-tile.
// Feature slab (32x512) staged once into LDS as f16; each wave = (msub, ng).
__global__ __launch_bounds__(256) void k_gemm1(const float* __restrict__ feat,
                                               const _Float16* __restrict__ W1h,
                                               const float* __restrict__ b1,
                                               _Float16* __restrict__ xhid) {
  __shared__ _Float16 smem[32 * ASTRIDE];   // 33,280 B

  const int tid = threadIdx.x;
  const int m0blk = blockIdx.x * 32;

  // ---- cooperative stage: 32 rows x 512 f32 -> f16 in LDS (4096 float4s) ----
  {
    const v4f* src = (const v4f*)(feat + (size_t)m0blk * NFEAT);
#pragma unroll
    for (int it = 0; it < 16; ++it) {
      int i4 = tid + it * 256;            // 0..4095
      int r  = i4 >> 7;                   // row 0..31   (128 float4 per row)
      int k  = (i4 & 127) * 4;
      v4f fv = src[i4];
      v4h hv;
      hv[0] = (_Float16)fv[0]; hv[1] = (_Float16)fv[1];
      hv[2] = (_Float16)fv[2]; hv[3] = (_Float16)fv[3];
      *(v4h*)(smem + r * ASTRIDE + k) = hv;
    }
  }
  __syncthreads();

  const int wave = tid >> 5;
  const int lane = tid & 31;
  const int ng   = wave >> 1;             // 0..3 : 64-col group
  const int msub = wave & 1;              // 0..1 : 16-row sub-tile
  const int r = lane & 15;
  const int h = lane >> 4;

  const _Float16* arow  = smem + (msub * 16 + r) * ASTRIDE;
  const _Float16* bbase = W1h + (size_t)(ng * 64 + r) * NFEAT;

  v8f acc[4] = {};
  for (int k0 = 0; k0 < NFEAT; k0 += 32) {
    int ka = k0 + h * 8;
    v8h alo = *(const v8h*)(arow + ka);        // ds_load_b128
    v8h ahi = *(const v8h*)(arow + ka + 16);
    v16h a = __builtin_shufflevector(alo, ahi, 0,1,2,3,4,5,6,7,8,9,10,11,12,13,14,15);
#pragma unroll
    for (int t = 0; t < 4; ++t) {
      const _Float16* brow = bbase + (size_t)t * 16 * NFEAT;
      v8h blo = *(const v8h*)(brow + ka);
      v8h bhi = *(const v8h*)(brow + ka + 16);
      v16h b = __builtin_shufflevector(blo, bhi, 0,1,2,3,4,5,6,7,8,9,10,11,12,13,14,15);
      acc[t] = __builtin_amdgcn_wmma_f32_16x16x32_f16(
          false, a, false, b, (short)0, acc[t], false, false);
    }
  }

  // D layout: lane, VGPR v -> C[m0 + 8*h + v][n0 + t*16 + r]
#pragma unroll
  for (int t = 0; t < 4; ++t) {
    int colc = ng * 64 + t * 16 + r;
    float bias = b1[colc];
#pragma unroll
    for (int v = 0; v < 8; ++v) {
      int rowc = m0blk + msub * 16 + h * 8 + v;
      float val = acc[t][v] + bias;
      xhid[(size_t)rowc * NHID + colc] = (_Float16)fmaxf(val, 0.0f);
    }
  }
}

__global__ __launch_bounds__(256) void k_gemm2(const _Float16* __restrict__ xhid,
                                               const _Float16* __restrict__ W2h,
                                               const float* __restrict__ b2,
                                               float* __restrict__ xout) {
  int wave = (int)((blockIdx.x * 256u + threadIdx.x) >> 5);
  int lane = threadIdx.x & 31;
  if (wave >= NN / 16) return;   // wave-uniform
  int r = lane & 15;
  int h = lane >> 4;
  const _Float16* arow = xhid + (size_t)(wave * 16 + r) * NHID;

  v8f acc[4] = {};
  for (int k0 = 0; k0 < NHID; k0 += 32) {
    int ka = k0 + h * 8;
    v8h alo = *(const v8h*)(arow + ka);
    v8h ahi = *(const v8h*)(arow + ka + 16);
    v16h a = __builtin_shufflevector(alo, ahi, 0,1,2,3,4,5,6,7,8,9,10,11,12,13,14,15);
#pragma unroll
    for (int t = 0; t < 4; ++t) {
      const _Float16* brow = W2h + (size_t)(t * 16 + r) * NHID;
      v8h blo = *(const v8h*)(brow + ka);
      v8h bhi = *(const v8h*)(brow + ka + 16);
      v16h b = __builtin_shufflevector(blo, bhi, 0,1,2,3,4,5,6,7,8,9,10,11,12,13,14,15);
      acc[t] = __builtin_amdgcn_wmma_f32_16x16x32_f16(
          false, a, false, b, (short)0, acc[t], false, false);
    }
  }
#pragma unroll
  for (int t = 0; t < 4; ++t) {
    int colc = t * 16 + r;
    float bias = b2[colc];
#pragma unroll
    for (int v = 0; v < 8; ++v) {
      int rowc = wave * 16 + h * 8 + v;
      xout[(size_t)rowc * NOUT + colc] = acc[t][v] + bias;
    }
  }
}

// ---------------- driver ----------------

extern "C" void kernel_launch(void* const* d_in, const int* in_sizes, int n_in,
                              void* d_out, int out_size, void* d_ws, size_t ws_size,
                              hipStream_t stream) {
  const float* feat = (const float*)d_in[0];
  const float* W1   = (const float*)d_in[1];
  const float* b1   = (const float*)d_in[2];
  const float* W2   = (const float*)d_in[3];
  const float* b2   = (const float*)d_in[4];
  const float* temp = (const float*)d_in[5];
  const int*   ei   = (const int*)d_in[6];
  const int E = in_sizes[6] / 2;          // 3,200,000 directed edges
  const int* row = ei;
  const int* col = ei + E;
  float* out = (float*)d_out;

  // workspace carve-up (~103.5 MB total)
  char* ws = (char*)d_ws;
  float*    deg   = (float*)ws;                        // N
  float*    dinv  = deg + NN;                          // N
  float*    coe   = dinv + NN;                         // 16
  _Float16* W1h   = (_Float16*)(coe + 16);             // NHID*NFEAT halves
  _Float16* W2h   = W1h + (size_t)NHID * NFEAT;        // NOUT*NHID halves
  _Float16* xhid  = W2h + (size_t)NOUT * NHID;         // N*NHID halves
  float*    xbuf  = (float*)(xhid + (size_t)NN * NHID);// N*NOUT f32  (Tx0)
  float*    t1buf = xbuf + (size_t)NN * NOUT;          // N*NOUT f32  (Tx1)

  const int NV = NN * NOUT;                 // 6.4M
  const int bN   = (NN + 255) / 256;
  const int bE   = (E + 255) / 256;
  const int bNV  = (NV + 255) / 256;
  const int bE32 = E / 8;                   // E*32/256, E divisible by 8
  const int bG1  = NN / 32;                 // 3125 blocks, 32 rows each
  const int bG2  = ((NN / 16) * 32 + 255) / 256;        // 782

  // degrees -> dinv, Chebyshev coefficients, weight pre-conversion
  k_zero<<<bN, 256, 0, stream>>>(deg, NN);
  k_deg<<<bE, 256, 0, stream>>>(row, deg, E);
  k_dinv<<<bN, 256, 0, stream>>>(deg, dinv, NN);
  k_coe<<<1, 32, 0, stream>>>(temp, coe);
  k_cvt_f16<<<(NHID * NFEAT + 255) / 256, 256, 0, stream>>>(W1, W1h, NHID * NFEAT);
  k_cvt_f16<<<(NOUT * NHID + 255) / 256, 256, 0, stream>>>(W2, W2h, NOUT * NHID);

  // MLP via WMMA
  k_gemm1<<<bG1, 256, 0, stream>>>(feat, W1h, b1, xhid);
  k_gemm2<<<bG2, 256, 0, stream>>>(xhid, W2h, b2, xbuf);   // xbuf = Tx0 = x

  // Tx1 = prop(x);  out = coe0/2 * Tx0 + coe1 * Tx1
  k_zero<<<bNV, 256, 0, stream>>>(t1buf, NV);
  k_scatter<<<bE32, 256, 0, stream>>>(row, col, dinv, xbuf, t1buf, 1.0f, E);
  k_combine0<<<bNV, 256, 0, stream>>>(xbuf, t1buf, coe, out, NV);

  // Chebyshev recursion: Tx2 = 2*prop(Tx1) - Tx0, done in-place in the Tx0 buffer
  float* Tx0 = xbuf;
  float* Tx1 = t1buf;
  for (int i = 2; i <= KORD; ++i) {
    k_negate<<<bNV, 256, 0, stream>>>(Tx0, NV);
    k_scatter<<<bE32, 256, 0, stream>>>(row, col, dinv, Tx1, Tx0, 2.0f, E);
    k_axpy<<<bNV, 256, 0, stream>>>(out, Tx0, coe, i, NV);
    float* t = Tx0; Tx0 = Tx1; Tx1 = t;
  }
}
